// RMGM_14972255993883
// MI455X (gfx1250) — matrix operand order repaired and checked by reading, not compiled, not run on hardware
//
#include <hip/hip_runtime.h>
#include <hip/hip_bf16.h>

// ---------------------------------------------------------------------------
// MI455X (gfx1250) AGCRN forward.
// ~215 GFLOP of GEMM mapped onto v_wmma_f32_16x16x32_bf16 (bf16 operands,
// fp32 accumulate). Both GEMM families stage their shared B panels into LDS
// with Tensor-Data-Mover descriptors (pad_enable chosen so LDS row strides
// are bank-conflict-free for ds_load_b128 column reads).
// ---------------------------------------------------------------------------

#define N_NODES 512
#define T_LEN   12
#define HID     64
#define BATCH   64
#define EDIM    10
#define IPAD0   96            // pad i=65 (D+H) -> 96 so 32-wide K-tiles don't straddle blocks
#define IPAD1   128           // i=128 (H+H) exact
#define CW0     (BATCH*IPAD0) // 6144
#define CW1     (BATCH*IPAD1) // 8192
#define KDIM    512           // support-GEMM reduction length
#define LDSROW  520           // 512 + 8 u16 pad (16B) -> 1040B stride, bank-conflict free

typedef unsigned short u16;
typedef __attribute__((ext_vector_type(16))) __bf16 bf16x16;
typedef __attribute__((ext_vector_type(8)))  float  f32x8;
typedef __attribute__((ext_vector_type(4)))  unsigned int u32x4;
typedef __attribute__((ext_vector_type(8)))  int i32x8;
typedef __attribute__((ext_vector_type(4)))  int i32x4;

union BfPack { uint4 q[2]; bf16x16 v; };

__device__ __forceinline__ u16 f32_to_bf16(float f) {
  unsigned u = __float_as_uint(f);
  return (u16)((u + 0x7FFFu + ((u >> 16) & 1u)) >> 16);   // RNE
}
__device__ __forceinline__ float bf16_to_f32(u16 h) {
  return __uint_as_float(((unsigned)h) << 16);
}

// A operand (16x32 bf16): lanes 0-15 row M=lane, K 0..15;
// lanes 16-31 same rows, K 16..31. Two 16B loads per lane.
__device__ __forceinline__ bf16x16 load_a16x32(const u16* __restrict__ A, int lda,
                                               int rowBase, int k0, int lane) {
  int row = rowBase + (lane & 15);
  int hlf = lane >> 4;
  const u16* p = A + (size_t)row * lda + k0 + hlf * 16;
  BfPack t;
  t.q[0] = *(const uint4*)(p);
  t.q[1] = *(const uint4*)(p + 8);
  return t.v;
}

// B operand from the LDS-staged support panel (row stride LDSROW u16).
__device__ __forceinline__ bf16x16 lds_b32x16(const u16* sB, int colBase, int k0, int lane) {
  int col  = colBase + (lane & 15);
  int koff = k0 + (lane >> 4) * 16;
  const u16* p = sB + col * LDSROW + koff;
  BfPack t;
  t.q[0] = *(const uint4*)(p);
  t.q[1] = *(const uint4*)(p + 8);
  return t.v;
}

// B operand from an LDS-staged weight panel with intra-row TDM padding:
// linear u16 index lin gets +8 u16 (16B) after every (1<<SHIFT) u16.
template <int SHIFT>
__device__ __forceinline__ bf16x16 lds_w32x16(const u16* sW, int col, int k0, int lane, int ktot) {
  int c    = col + (lane & 15);
  int koff = k0 + (lane >> 4) * 16;
  int lin0 = c * ktot + koff;
  int lin1 = lin0 + 8;
  BfPack t;
  t.q[0] = *(const uint4*)(sW + lin0 + ((lin0 >> SHIFT) << 3));
  t.q[1] = *(const uint4*)(sW + lin1 + ((lin1 >> SHIFT) << 3));
  return t.v;
}

// ---------------------------------------------------------------------------
// Tensor Data Mover: 2D tile (rows x rowElems u16) global -> LDS.
// w0flags carries data_size + pad configuration.
// ---------------------------------------------------------------------------
__device__ __forceinline__ void tdm_load_2d(unsigned ldsOff, const void* gptr,
                                            unsigned rowElems, unsigned rows,
                                            unsigned strideElems, unsigned w0flags) {
  unsigned long long ga = (unsigned long long)gptr;
  u32x4 g0;
  g0[0] = 1u;                                            // count=1, user descriptor
  g0[1] = ldsOff;                                        // lds_addr
  g0[2] = (unsigned)ga;                                  // global_addr[31:0]
  g0[3] = ((unsigned)(ga >> 32) & 0x01FFFFFFu) | (2u << 30);  // addr[56:32] | type=2
  i32x8 g1;
  g1[0] = (int)w0flags;                                  // wg_mask=0 | data_size | pad cfg
  g1[1] = (int)((rowElems & 0xFFFFu) << 16);             // tensor_dim0[15:0]
  g1[2] = (int)((rowElems >> 16) | ((rows & 0xFFFFu) << 16));   // td0 hi | td1 lo
  g1[3] = (int)((rows >> 16) | ((rowElems & 0xFFFFu) << 16));   // td1 hi | tile_dim0
  g1[4] = (int)(rows & 0xFFFFu);                         // tile_dim1 (tile_dim2 = 0)
  g1[5] = (int)strideElems;                              // tensor_dim0_stride[31:0]
  g1[6] = 0;
  g1[7] = 0;
  i32x4 z4 = {0, 0, 0, 0};
#if __clang_major__ >= 23
  i32x8 z8 = {0, 0, 0, 0, 0, 0, 0, 0};
  __builtin_amdgcn_tensor_load_to_lds(g0, g1, z4, z4, z8, 0);
#else
  __builtin_amdgcn_tensor_load_to_lds(g0, g1, z4, z4, 0);
#endif
}

// data_size=2B (code 1); pad_enable; pad_amount code 3 (4 DWORDs = 16B);
// interval code 7 = 256 DWORDs (one 512-u16 row), 6 = 128 DWORDs, 4 = 32 DWORDs.
#define TDM_FLAGS(ivcode) ((1u << 16) | (1u << 20) | ((unsigned)(ivcode) << 22) | (3u << 25))

// ---------------------------------------------------------------------------
// Graph construction: adj is exactly y_hard (straight-through), i.e. binary.
// ---------------------------------------------------------------------------
__global__ void k_adj(const float* __restrict__ E, const float* __restrict__ gn,
                      unsigned char* __restrict__ adj, int* __restrict__ deg,
                      int* __restrict__ maxdeg, int* __restrict__ anyedge) {
  __shared__ float en[16];
  __shared__ int sred[256];
  int n = blockIdx.x, tid = threadIdx.x;
  if (tid < EDIM) en[tid] = E[n * EDIM + tid];
  __syncthreads();
  float nn = 0.f;
  for (int d = 0; d < EDIM; d++) nn += en[d] * en[d];
  nn = sqrtf(nn);
  int cnt = 0;
  for (int m = tid; m < N_NODES; m += 256) {
    float dot = 0.f, m2 = 0.f;
    for (int d = 0; d < EDIM; d++) { float e = E[m * EDIM + d]; dot += en[d] * e; m2 += e * e; }
    float lg = dot / (nn * sqrtf(m2));
    lg = 0.5f * (lg + 1.f);
    size_t gi = ((size_t)n * N_NODES + m) * 2;
    int a = (n != m) && (lg + gn[gi] >= (1.f - lg) + gn[gi + 1]);  // argmax ties -> idx 0
    adj[(size_t)n * N_NODES + m] = (unsigned char)a;
    cnt += a;
  }
  sred[tid] = cnt; __syncthreads();
  for (int s = 128; s > 0; s >>= 1) { if (tid < s) sred[tid] += sred[tid + s]; __syncthreads(); }
  if (tid == 0) {
    deg[n] = sred[0];
    atomicMax(maxdeg, sred[0]);
    if (sred[0] > 0) atomicOr(anyedge, 1);
  }
}

// tilde = 2*(diag(d)-adj)/lmax - I, lmax = maxdeg + (anyedge?1:0)
__global__ void k_tilde(const unsigned char* __restrict__ adj, const int* __restrict__ deg,
                        const int* __restrict__ maxdeg, const int* __restrict__ anyedge,
                        u16* __restrict__ tilde_bf) {
  int idx = blockIdx.x * 256 + threadIdx.x;  // 512*512
  int m = idx & (N_NODES - 1), n = idx >> 9;
  float lmax = (float)(*maxdeg) + ((*anyedge) ? 1.f : 0.f);
  if (lmax == 0.f) lmax = 1.f;
  float L = (n == m ? (float)deg[n] : 0.f) - (float)adj[idx];
  float v = 2.f * L / lmax - (n == m ? 1.f : 0.f);
  tilde_bf[idx] = f32_to_bf16(v);
}

// ---------------------------------------------------------------------------
// Per-node weight pools: W_bt[n][o][k*ipad+i] = sum_d E[n,d]*Wp[d,k,i,o]
// (bf16, pre-transposed for the WMMA B operand; i>=isize zero-padded).
// ---------------------------------------------------------------------------
__global__ void k_make_w(const float* __restrict__ E, const float* __restrict__ Wp,
                         u16* __restrict__ wbt, int isize, int osz, int ipad) {
  size_t idx = (size_t)blockIdx.x * 256 + threadIdx.x;  // N*osz*(2*ipad)
  int ktot = 2 * ipad;
  int kk = (int)(idx % ktot);
  size_t r = idx / ktot;
  int oo = (int)(r % osz);
  int n  = (int)(r / osz);
  int k = kk / ipad, i = kk % ipad;
  float v = 0.f;
  if (i < isize) {
    for (int d = 0; d < EDIM; d++)
      v += E[n * EDIM + d] * Wp[(((size_t)d * 2 + k) * isize + i) * osz + oo];
  }
  wbt[idx] = f32_to_bf16(v);
}

__global__ void k_make_b(const float* __restrict__ E, const float* __restrict__ bp,
                         float* __restrict__ bout, int osz) {
  int idx = blockIdx.x * 256 + threadIdx.x;  // N*osz
  int oo = idx % osz, n = idx / osz;
  float v = 0.f;
  for (int d = 0; d < EDIM; d++) v += E[n * EDIM + d] * bp[d * osz + oo];
  bout[idx] = v;
}

// ---------------------------------------------------------------------------
// Pack concat([x, (z*)h]) into BOTH operand layouts in one pass:
//   arow [n][b][ipad] bf16 (A operand of per-node GEMM, identity support)
//   acol [(b*ipad+i)][m] bf16 (pre-transposed B operand of tilde-GEMM)
// ---------------------------------------------------------------------------
__global__ void k_pack(const float* __restrict__ gts, int t,
                       const u16* __restrict__ xbf, int xdim,
                       const float* __restrict__ h, const float* __restrict__ zr,
                       u16* __restrict__ arow, u16* __restrict__ acol, int ipad) {
  size_t idx = (size_t)blockIdx.x * 256 + threadIdx.x;  // N*B*ipad
  int i = (int)(idx % ipad);
  size_t r = idx / ipad;
  int b = (int)(r % BATCH);
  int m = (int)(r / BATCH);
  float v = 0.f;
  if (i < xdim) {
    v = gts ? gts[((size_t)b * N_NODES + m) * T_LEN + t]                 // layer0: D=1
            : bf16_to_f32(xbf[((size_t)m * BATCH + b) * HID + i]);       // layer1: x = h0
  } else if (i < xdim + HID) {
    int j = i - xdim;
    float hv = h[((size_t)m * BATCH + b) * HID + j];
    if (zr) hv *= zr[((size_t)m * BATCH + b) * 128 + j];                 // z*h for update path
    v = hv;
  }
  u16 bf = f32_to_bf16(v);
  arow[idx] = bf;
  acol[((size_t)b * ipad + i) * N_NODES + m] = bf;
}

// ---------------------------------------------------------------------------
// Support GEMM: C[512][Cw] = tilde[512][512] x acol^T, bf16 out.
// Block tile 256x64: the 64x512 B-panel is TDM-staged into padded LDS once,
// then 8 waves each compute a 32x64 strip (2 A-tiles x 4 B-tiles = 8 WMMAs
// per K-step, loads batched ahead of the WMMA burst).
// ---------------------------------------------------------------------------
__global__ void k_gemm_bf16(const u16* __restrict__ A, const u16* __restrict__ Bt,
                            u16* __restrict__ C, int lda, int ldb, int ldc) {
  __shared__ u16 sB[64 * LDSROW];
  int wave = threadIdx.x >> 5, lane = threadIdx.x & 31;
  int rowBase = blockIdx.x * 256 + wave * 32;
  int colBase = blockIdx.y * 64;

  if (threadIdx.x < 32) {   // one wave issues the DMA descriptor
    tdm_load_2d((unsigned)(size_t)&sB[0], Bt + (size_t)colBase * ldb,
                KDIM, 64, (unsigned)ldb, TDM_FLAGS(7));   // pad 16B per 256-DWORD row
    __builtin_amdgcn_s_wait_tensorcnt(0);
  }
  __syncthreads();

  f32x8 acc[2][4] = {};
  for (int k0 = 0; k0 < KDIM; k0 += 32) {
    bf16x16 a0 = load_a16x32(A, lda, rowBase, k0, lane);
    bf16x16 a1 = load_a16x32(A, lda, rowBase + 16, k0, lane);
    if (k0 + 32 < KDIM)
      __builtin_prefetch(A + (size_t)(rowBase + (lane & 15)) * lda + k0 + 32, 0, 1);
    bf16x16 b[4];
#pragma unroll
    for (int s = 0; s < 4; s++) b[s] = lds_b32x16(sB, s * 16, k0, lane);
#pragma unroll
    for (int s = 0; s < 4; s++)
      acc[0][s] = __builtin_amdgcn_wmma_f32_16x16x32_bf16(false, a0, false, b[s],
                                                          (short)0, acc[0][s], false, false);
#pragma unroll
    for (int s = 0; s < 4; s++)
      acc[1][s] = __builtin_amdgcn_wmma_f32_16x16x32_bf16(false, a1, false, b[s],
                                                          (short)0, acc[1][s], false, false);
  }

  int hlf = lane >> 4, cIn = lane & 15;
#pragma unroll
  for (int g = 0; g < 2; g++) {
#pragma unroll
    for (int s = 0; s < 4; s++) {
      int col = colBase + s * 16 + cIn;
#pragma unroll
      for (int j = 0; j < 8; j++) {
        int row = rowBase + g * 16 + j + hlf * 8;
        C[(size_t)row * ldc + col] = f32_to_bf16(acc[g][s][j]);
      }
    }
  }
}

// ---------------------------------------------------------------------------
// Per-node GEMM with fused bias+activation epilogue. The node's weight panel
// (OSZ x KTOT bf16) is TDM-staged into LDS with intra-row padding so the
// ds_load_b128 column reads are bank-conflict-free; A tiles stay in global.
// node n: Out[b,o] = act( [arow_n | xg1_n][64 x KTOT] x W_n[KTOT x OSZ] + bias_n )
// ---------------------------------------------------------------------------
template <int OSZ, int IPAD, int KTHALF, int ACT>
__global__ void k_gemm_node(const u16* __restrict__ Arow, const u16* __restrict__ Xg1,
                            const u16* __restrict__ Wbt, const float* __restrict__ bias,
                            float* __restrict__ Out) {
  constexpr int KTOT  = 2 * IPAD;
  constexpr int PWN   = (OSZ / 16) / 2;            // subtiles/wave: 4 (OSZ=128) or 2 (OSZ=64)
  constexpr int SHIFT = (KTOT == 256) ? 8 : 6;     // pad every 256 / 64 u16
  constexpr unsigned IVCODE = (KTOT == 256) ? 6u : 4u;  // 128 / 32 DWORD interval
  constexpr int LDSW  = OSZ * KTOT + ((OSZ * KTOT) >> SHIFT) * 8;
  __shared__ u16 sW[LDSW];
  int n = blockIdx.x;
  int wave = threadIdx.x >> 5, lane = threadIdx.x & 31;
  int mt = wave >> 1;                              // 4 M-tiles of the 64 batch rows
  int ntBase = (wave & 1) * PWN;
  const u16* a0 = Arow + (size_t)n * BATCH * IPAD; // identity support (k=0)
  const u16* a1 = Xg1  + (size_t)n * BATCH * IPAD; // tilde support   (k=1)

  if (threadIdx.x < 32) {
    tdm_load_2d((unsigned)(size_t)&sW[0], Wbt + (size_t)n * OSZ * KTOT,
                KTOT, OSZ, KTOT, TDM_FLAGS(IVCODE));
    __builtin_amdgcn_s_wait_tensorcnt(0);
  }
  __syncthreads();

  f32x8 acc[PWN] = {};
#pragma unroll
  for (int kk = 0; kk < 2 * KTHALF; kk++) {
    const u16* src = (kk < KTHALF) ? a0 : a1;
    int kLoc = ((kk < KTHALF) ? kk : kk - KTHALF) * 32;
    bf16x16 a = load_a16x32(src, IPAD, mt * 16, kLoc, lane);
    bf16x16 b[PWN];
#pragma unroll
    for (int s = 0; s < PWN; s++)
      b[s] = lds_w32x16<SHIFT>(sW, (ntBase + s) * 16, kk * 32, lane, KTOT);
#pragma unroll
    for (int s = 0; s < PWN; s++)
      acc[s] = __builtin_amdgcn_wmma_f32_16x16x32_bf16(false, a, false, b[s],
                                                       (short)0, acc[s], false, false);
  }
  int hlf = lane >> 4, cIn = lane & 15;
#pragma unroll
  for (int s = 0; s < PWN; s++) {
    int oo = (ntBase + s) * 16 + cIn;
    float bv = bias[n * OSZ + oo];
#pragma unroll
    for (int j = 0; j < 8; j++) {
      int bi = mt * 16 + j + hlf * 8;
      float v = acc[s][j] + bv;
      v = (ACT == 0) ? (1.f / (1.f + __expf(-v))) : tanhf(v);
      Out[((size_t)n * BATCH + bi) * OSZ + oo] = v;
    }
  }
}

// h_new = r*h + (1-r)*hc  (z,r already sigmoided; hc already tanh'd)
__global__ void k_hupd(const float* __restrict__ zr, const float* __restrict__ hc,
                       float* __restrict__ h, u16* __restrict__ hbf) {
  int idx = blockIdx.x * 256 + threadIdx.x;  // N*B*H
  int j = idx & (HID - 1);
  size_t r = (size_t)idx >> 6;               // m*BATCH + b
  float rr  = zr[r * 128 + HID + j];
  float hn  = rr * h[idx] + (1.f - rr) * hc[idx];
  h[idx]   = hn;
  hbf[idx] = f32_to_bf16(hn);
}

// end_conv: out[b,t,n,0] = sum_h h1[n][b][h]*cw[t][h] + cb[t]
__global__ void k_endconv(const float* __restrict__ h1, const float* __restrict__ cw,
                          const float* __restrict__ cb, float* __restrict__ out) {
  int idx = blockIdx.x * 256 + threadIdx.x;  // B*T*N
  int n = idx & (N_NODES - 1);
  int r = idx >> 9;
  int t = r % T_LEN, b = r / T_LEN;
  float s = cb[t];
  const float* hp = h1 + ((size_t)n * BATCH + b) * HID;
  for (int hh = 0; hh < HID; hh++) s += hp[hh] * cw[t * HID + hh];
  out[((size_t)b * T_LEN + t) * N_NODES + n] = s;
}

// ---------------------------------------------------------------------------
extern "C" void kernel_launch(void* const* d_in, const int* in_sizes, int n_in,
                              void* d_out, int out_size, void* d_ws, size_t ws_size,
                              hipStream_t stream) {
  (void)in_sizes; (void)n_in; (void)out_size; (void)ws_size;
  const float* gts = (const float*)d_in[0];
  const float* gn  = (const float*)d_in[1];
  const float* E   = (const float*)d_in[2];
  const float* gw0 = (const float*)d_in[3];
  const float* gb0 = (const float*)d_in[4];
  const float* uw0 = (const float*)d_in[5];
  const float* ub0 = (const float*)d_in[6];
  const float* gw1 = (const float*)d_in[7];
  const float* gb1 = (const float*)d_in[8];
  const float* uw1 = (const float*)d_in[9];
  const float* ub1 = (const float*)d_in[10];
  const float* cw  = (const float*)d_in[11];
  const float* cb  = (const float*)d_in[12];
  float* out = (float*)d_out;

  char* p = (char*)d_ws;
  auto alloc = [&](size_t bytes) -> char* {
    char* q = p; p += (bytes + 255) & ~(size_t)255; return q;
  };
  unsigned char* adj = (unsigned char*)alloc((size_t)N_NODES * N_NODES);
  int* deg     = (int*)alloc(N_NODES * 4 + 256);
  int* maxdeg  = deg + N_NODES;
  int* anyedge = deg + N_NODES + 1;
  u16* tilde = (u16*)alloc((size_t)N_NODES * N_NODES * 2);
  u16* wbt0g = (u16*)alloc((size_t)N_NODES * 128 * (2 * IPAD0) * 2);
  u16* wbt0u = (u16*)alloc((size_t)N_NODES *  64 * (2 * IPAD0) * 2);
  u16* wbt1g = (u16*)alloc((size_t)N_NODES * 128 * (2 * IPAD1) * 2);
  u16* wbt1u = (u16*)alloc((size_t)N_NODES *  64 * (2 * IPAD1) * 2);
  float* b0g = (float*)alloc(N_NODES * 128 * 4);
  float* b0u = (float*)alloc(N_NODES *  64 * 4);
  float* b1g = (float*)alloc(N_NODES * 128 * 4);
  float* b1u = (float*)alloc(N_NODES *  64 * 4);
  u16* arow = (u16*)alloc((size_t)N_NODES * BATCH * IPAD1 * 2);
  u16* acol = (u16*)alloc((size_t)CW1 * N_NODES * 2);
  u16* xg1  = (u16*)alloc((size_t)N_NODES * CW1 * 2);
  float* zr = (float*)alloc((size_t)N_NODES * BATCH * 128 * 4);
  float* hc = (float*)alloc((size_t)N_NODES * BATCH * HID * 4);
  float* h0 = (float*)alloc((size_t)N_NODES * BATCH * HID * 4);
  float* h1 = (float*)alloc((size_t)N_NODES * BATCH * HID * 4);
  u16* hbf0 = (u16*)alloc((size_t)N_NODES * BATCH * HID * 2);
  u16* hbf1 = (u16*)alloc((size_t)N_NODES * BATCH * HID * 2);

  (void)hipMemsetAsync(maxdeg, 0, 8, stream);
  (void)hipMemsetAsync(h0, 0, (size_t)N_NODES * BATCH * HID * 4, stream);
  (void)hipMemsetAsync(h1, 0, (size_t)N_NODES * BATCH * HID * 4, stream);

  k_adj<<<N_NODES, 256, 0, stream>>>(E, gn, adj, deg, maxdeg, anyedge);
  k_tilde<<<(N_NODES * N_NODES) / 256, 256, 0, stream>>>(adj, deg, maxdeg, anyedge, tilde);

  k_make_w<<<(N_NODES * 128 * 2 * IPAD0) / 256, 256, 0, stream>>>(E, gw0, wbt0g, 65, 128, IPAD0);
  k_make_w<<<(N_NODES *  64 * 2 * IPAD0) / 256, 256, 0, stream>>>(E, uw0, wbt0u, 65,  64, IPAD0);
  k_make_w<<<(N_NODES * 128 * 2 * IPAD1) / 256, 256, 0, stream>>>(E, gw1, wbt1g, 128, 128, IPAD1);
  k_make_w<<<(N_NODES *  64 * 2 * IPAD1) / 256, 256, 0, stream>>>(E, uw1, wbt1u, 128,  64, IPAD1);
  k_make_b<<<(N_NODES * 128) / 256, 256, 0, stream>>>(E, gb0, b0g, 128);
  k_make_b<<<(N_NODES *  64) / 256, 256, 0, stream>>>(E, ub0, b0u,  64);
  k_make_b<<<(N_NODES * 128) / 256, 256, 0, stream>>>(E, gb1, b1g, 128);
  k_make_b<<<(N_NODES *  64) / 256, 256, 0, stream>>>(E, ub1, b1u,  64);

  const int packB0 = (N_NODES * BATCH * IPAD0) / 256;
  const int packB1 = (N_NODES * BATCH * IPAD1) / 256;
  dim3 gridS0(N_NODES / 256, CW0 / 64), gridS1(N_NODES / 256, CW1 / 64);
  const int hupdB = (N_NODES * BATCH * HID) / 256;

  for (int t = 0; t < T_LEN; t++) {
    // ---- layer 0 (x = gts[:, :, t, 0]) ----
    k_pack<<<packB0, 256, 0, stream>>>(gts, t, nullptr, 1, h0, nullptr, arow, acol, IPAD0);
    k_gemm_bf16<<<gridS0, 256, 0, stream>>>(tilde, acol, xg1, N_NODES, N_NODES, CW0);
    k_gemm_node<128, IPAD0, 3, 0><<<N_NODES, 256, 0, stream>>>(arow, xg1, wbt0g, b0g, zr);
    k_pack<<<packB0, 256, 0, stream>>>(gts, t, nullptr, 1, h0, zr, arow, acol, IPAD0);
    k_gemm_bf16<<<gridS0, 256, 0, stream>>>(tilde, acol, xg1, N_NODES, N_NODES, CW0);
    k_gemm_node<64, IPAD0, 3, 1><<<N_NODES, 256, 0, stream>>>(arow, xg1, wbt0u, b0u, hc);
    k_hupd<<<hupdB, 256, 0, stream>>>(zr, hc, h0, hbf0);
    // ---- layer 1 (x = h0 of this step) ----
    k_pack<<<packB1, 256, 0, stream>>>(nullptr, t, hbf0, HID, h1, nullptr, arow, acol, IPAD1);
    k_gemm_bf16<<<gridS1, 256, 0, stream>>>(tilde, acol, xg1, N_NODES, N_NODES, CW1);
    k_gemm_node<128, IPAD1, 4, 0><<<N_NODES, 256, 0, stream>>>(arow, xg1, wbt1g, b1g, zr);
    k_pack<<<packB1, 256, 0, stream>>>(nullptr, t, hbf0, HID, h1, zr, arow, acol, IPAD1);
    k_gemm_bf16<<<gridS1, 256, 0, stream>>>(tilde, acol, xg1, N_NODES, N_NODES, CW1);
    k_gemm_node<64, IPAD1, 4, 1><<<N_NODES, 256, 0, stream>>>(arow, xg1, wbt1u, b1u, hc);
    k_hupd<<<hupdB, 256, 0, stream>>>(zr, hc, h1, hbf1);
  }
  k_endconv<<<(BATCH * T_LEN * N_NODES) / 256, 256, 0, stream>>>(h1, cw, cb, out);
}
